// CortexVIII_STMM_41549513621992
// MI455X (gfx1250) — matrix operand order repaired and checked by reading, not compiled, not run on hardware
//
#include <hip/hip_runtime.h>

// ---------------- model constants ----------------
#define BB   2
#define TT   1024
#define DDIM 768
#define VV   32000
#define LL   4
#define HH   12
#define DHH  64      // == DM
#define DFFC 3072
#define WINC 256
#define ES   4       // entity slots
#define KCB  64      // codebook entries
#define NTG  64      // tags

typedef __attribute__((ext_vector_type(4)))  float    v4f;
typedef __attribute__((ext_vector_type(8)))  float    v8f;
typedef __attribute__((ext_vector_type(8)))  _Float16 v8h;
typedef __attribute__((ext_vector_type(16))) _Float16 v16h;
typedef int v4i_b128 __attribute__((vector_size(16)));   // matches builtin param pointee

// ---------------- CDNA5 async global->LDS copy (guarded) ----------------
#if defined(__HIP_DEVICE_COMPILE__) && \
    __has_builtin(__builtin_amdgcn_global_load_async_to_lds_b128) && \
    __has_builtin(__builtin_amdgcn_s_wait_asynccnt)
#define USE_ASYNC_LDS 1
#else
#define USE_ASYNC_LDS 0
#endif

__device__ __forceinline__ void lds_copy_b128(const float* __restrict__ g, float* l) {
#if USE_ASYNC_LDS
    typedef __attribute__((address_space(1))) v4i_b128* gp_t;   // global int4*
    typedef __attribute__((address_space(3))) v4i_b128* lp_t;   // LDS int4*
    __builtin_amdgcn_global_load_async_to_lds_b128((gp_t)(void*)g, (lp_t)(void*)l, 0, 0);
#else
    *(v4f*)l = *(const v4f*)g;
#endif
}
__device__ __forceinline__ void lds_copy_wait() {
#if USE_ASYNC_LDS
    __builtin_amdgcn_s_wait_asynccnt(0);
#endif
    __syncthreads();
}

// =================================================================
// WMMA GEMM:  C[M,N] = A[M,K] @ W[N,K]^T   (fp32 in, fused f16 cvt,
// fp32 out). block tile 128x128, K-step 32, 256 threads = 8 waves.
// wave grid 4x2 -> each wave 32x64 of C (2x4 subtiles of 16x16)
// =================================================================
__global__ __launch_bounds__(256)
void kern_gemm(const float* __restrict__ A,
               const float* __restrict__ W,
               float* __restrict__ C,
               const float* __restrict__ Res,   // optional residual (may alias C)
               int M, int N, int Kd, int epi)   // epi: 0=none, 1=sigmoid
{
    __shared__ alignas(64) _Float16 As[128][32];
    __shared__ alignas(64) _Float16 Bs[128][32];

    const int tid  = threadIdx.x;
    const int bm   = blockIdx.y * 128;
    const int bn   = blockIdx.x * 128;
    const int wave = tid >> 5;
    const int lane = tid & 31;
    const int wm   = (wave & 3) * 32;     // wave row offset in tile
    const int wn   = (wave >> 2) * 64;    // wave col offset in tile
    const int lm   = lane & 15;
    const int kh   = lane >> 4;

    v8f acc[2][4];
#pragma unroll
    for (int i = 0; i < 2; i++)
#pragma unroll
        for (int j = 0; j < 4; j++)
#pragma unroll
            for (int e = 0; e < 8; e++) acc[i][j][e] = 0.0f;

    // staging: 256 threads cover 128 rows x 32 cols, 16 f32 each
    const int lr = tid >> 1;
    const int lc = (tid & 1) * 16;

    for (int k0 = 0; k0 < Kd; k0 += 32) {
        const v4f* ga = (const v4f*)(A + (size_t)(bm + lr) * Kd + k0 + lc);
        const v4f* gb = (const v4f*)(W + (size_t)(bn + lr) * Kd + k0 + lc);
        v4f a0 = ga[0], a1 = ga[1], a2 = ga[2], a3 = ga[3];
        v4f b0 = gb[0], b1 = gb[1], b2 = gb[2], b3 = gb[3];
        v16h ah, bh;
#pragma unroll
        for (int e = 0; e < 4; e++) {
            ah[e]      = (_Float16)a0[e];  ah[4 + e]  = (_Float16)a1[e];
            ah[8 + e]  = (_Float16)a2[e];  ah[12 + e] = (_Float16)a3[e];
            bh[e]      = (_Float16)b0[e];  bh[4 + e]  = (_Float16)b1[e];
            bh[8 + e]  = (_Float16)b2[e];  bh[12 + e] = (_Float16)b3[e];
        }
        *(v16h*)&As[lr][lc] = ah;
        *(v16h*)&Bs[lr][lc] = bh;
        __syncthreads();

        // A fragments (ISA 16-bit A 16x32 layout): runs [kh*8,+8) and [16+kh*8,+8)
        v16h afrag[2];
#pragma unroll
        for (int i = 0; i < 2; i++) {
            const v8h r0 = *(const v8h*)&As[wm + i * 16 + lm][kh * 8];
            const v8h r1 = *(const v8h*)&As[wm + i * 16 + lm][16 + kh * 8];
            v16h a;
#pragma unroll
            for (int e = 0; e < 8; e++) { a[e] = r0[e]; a[8 + e] = r1[e]; }
            afrag[i] = a;
        }
#pragma unroll
        for (int j = 0; j < 4; j++) {
            // B fragment: lane holds column n=lm, runs [kh*8,+8) and [16+kh*8,+8)
            const v8h q0 = *(const v8h*)&Bs[wn + j * 16 + lm][kh * 8];
            const v8h q1 = *(const v8h*)&Bs[wn + j * 16 + lm][16 + kh * 8];
            v16h b;
#pragma unroll
            for (int e = 0; e < 8; e++) { b[e] = q0[e]; b[8 + e] = q1[e]; }
#pragma unroll
            for (int i = 0; i < 2; i++)
                acc[i][j] = __builtin_amdgcn_wmma_f32_16x16x32_f16(
                    false, afrag[i], false, b, (short)0, acc[i][j], false, false);
        }
        __syncthreads();
    }

    // C layout: VGPR e -> row kh*8+e, col lm
#pragma unroll
    for (int i = 0; i < 2; i++)
#pragma unroll
        for (int j = 0; j < 4; j++) {
            const int rbase = bm + wm + i * 16 + kh * 8;
            const int col   = bn + wn + j * 16 + lm;
#pragma unroll
            for (int e = 0; e < 8; e++) {
                float v = acc[i][j][e];
                size_t idx = (size_t)(rbase + e) * N + col;
                if (Res) v += Res[idx];
                if (epi == 1) v = 1.0f / (1.0f + __expf(-v));
                C[idx] = v;
            }
        }
}

// ---------------- elementwise ----------------
__global__ void kern_zero(float* p, size_t n) {
    size_t i = blockIdx.x * (size_t)blockDim.x + threadIdx.x;
    size_t s = gridDim.x * (size_t)blockDim.x;
    for (; i < n; i += s) p[i] = 0.0f;
}
__global__ void kern_add(float* __restrict__ x, const float* __restrict__ y, size_t n) {
    size_t i = blockIdx.x * (size_t)blockDim.x + threadIdx.x;
    size_t s = gridDim.x * (size_t)blockDim.x;
    for (; i < n; i += s) x[i] += y[i];
}
__global__ void kern_mix(float* __restrict__ loc, const float* __restrict__ gate,
                         const float* __restrict__ glob, size_t n) {
    size_t i = blockIdx.x * (size_t)blockDim.x + threadIdx.x;
    size_t s = gridDim.x * (size_t)blockDim.x;
    for (; i < n; i += s) { float g = gate[i]; loc[i] = g * loc[i] + (1.0f - g) * glob[i]; }
}
__global__ void kern_silu_mul(float* __restrict__ g, const float* __restrict__ u, size_t n) {
    size_t i = blockIdx.x * (size_t)blockDim.x + threadIdx.x;
    size_t s = gridDim.x * (size_t)blockDim.x;
    for (; i < n; i += s) { float a = g[i]; g[i] = (a / (1.0f + __expf(-a))) * u[i]; }
}
__global__ void kern_embed(const float* __restrict__ emb, const int* __restrict__ tok,
                           float* __restrict__ x, size_t n) {
    size_t i = blockIdx.x * (size_t)blockDim.x + threadIdx.x;
    size_t s = gridDim.x * (size_t)blockDim.x;
    for (; i < n; i += s) {
        size_t bt = i / DDIM, d = i % DDIM;
        x[i] = emb[(size_t)tok[bt] * DDIM + d];
    }
}

// ---------------- RMSNorm: one block per row ----------------
__global__ __launch_bounds__(256)
void kern_rms(const float* __restrict__ x, const float* __restrict__ w,
              float* __restrict__ y) {
    __shared__ float red[256];
    size_t row = blockIdx.x;
    const float* xr = x + row * DDIM;
    float s = 0.f;
    for (int d = threadIdx.x; d < DDIM; d += 256) { float v = xr[d]; s += v * v; }
    red[threadIdx.x] = s; __syncthreads();
    for (int o = 128; o > 0; o >>= 1) {
        if (threadIdx.x < (unsigned)o) red[threadIdx.x] += red[threadIdx.x + o];
        __syncthreads();
    }
    float inv = rsqrtf(red[0] / (float)DDIM + 1e-6f);
    for (int d = threadIdx.x; d < DDIM; d += 256) y[row * DDIM + d] = xr[d] * inv * w[d];
}

// ---------------- L2-normalize 64-length head vectors in place ----------------
__global__ void kern_l2n(float* __restrict__ buf, size_t nvec) {
    size_t i = blockIdx.x * (size_t)blockDim.x + threadIdx.x;
    size_t st = gridDim.x * (size_t)blockDim.x;
    for (; i < nvec; i += st) {
        float* v = buf + i * DHH;
        float s = 0.f;
#pragma unroll 8
        for (int d = 0; d < DHH; d++) s += v[d] * v[d];
        float inv = 1.0f / fmaxf(sqrtf(s), 1e-12f);
#pragma unroll 8
        for (int d = 0; d < DHH; d++) v[d] *= inv;
    }
}

// ---------------- beta = sigmoid(h2 @ betaW^T), stored (B,H,T) ----------------
__global__ __launch_bounds__(256)
void kern_beta(const float* __restrict__ h2, const float* __restrict__ bw,
               float* __restrict__ beta) {
    __shared__ float red[256];
    size_t bt = blockIdx.x;
    int b = (int)(bt / TT), t = (int)(bt % TT);
    const float* xr = h2 + bt * DDIM;
    for (int h = 0; h < HH; h++) {
        float s = 0.f;
        for (int d = threadIdx.x; d < DDIM; d += 256) s += xr[d] * bw[h * DDIM + d];
        red[threadIdx.x] = s; __syncthreads();
        for (int o = 128; o > 0; o >>= 1) {
            if (threadIdx.x < (unsigned)o) red[threadIdx.x] += red[threadIdx.x + o];
            __syncthreads();
        }
        if (threadIdx.x == 0)
            beta[((size_t)b * HH + h) * TT + t] = 1.0f / (1.0f + __expf(-red[0]));
        __syncthreads();
    }
}

// ---------------- cum = cumsum_t log(1-beta+1e-8) per (b,h) ----------------
__global__ void kern_cumsum(const float* __restrict__ beta, float* __restrict__ cum) {
    int i = threadIdx.x;
    if (i < BB * HH) {
        const float* br = beta + (size_t)i * TT;
        float* cr = cum + (size_t)i * TT;
        float s = 0.f;
        for (int t = 0; t < TT; t++) { s += logf(1.0f - br[t] + 1e-8f); cr[t] = s; }
    }
}

// ---------------- sliding-window attention (online softmax) ----------------
// grid (T/64, H, B), block 64: one thread per query
__global__ __launch_bounds__(64)
void kern_swa(const float* __restrict__ qkv, float* __restrict__ o) {
    __shared__ float Ks[64][64];
    __shared__ float Vs[64][64];
    const int qt = blockIdx.x, h = blockIdx.y, b = blockIdx.z;
    const int q = qt * 64 + threadIdx.x;
    const size_t rs = 3 * HH * DHH;

    float qv[DHH];
#pragma unroll 8
    for (int d = 0; d < DHH; d++)
        qv[d] = qkv[((size_t)b * TT + q) * rs + h * DHH + d];

    float m = -1e30f, l = 0.f;
    float accv[DHH];
#pragma unroll 8
    for (int d = 0; d < DHH; d++) accv[d] = 0.f;

    for (int cs = qt * 64 - WINC; cs <= qt * 64; cs += 64) {
        if (cs + 63 < 0) continue;
        // stage K/V chunk via async b128 copies (valid rows only; compute masks the rest)
        for (int u = threadIdx.x; u < 64 * 16; u += 64) {
            int rr = u >> 4, dd = (u & 15) * 4;
            int t = cs + rr;
            if (t >= 0 && t < TT) {
                size_t base = ((size_t)b * TT + t) * rs + h * DHH + dd;
                lds_copy_b128(&qkv[base + HH * DHH],     &Ks[rr][dd]);
                lds_copy_b128(&qkv[base + 2 * HH * DHH], &Vs[rr][dd]);
            }
        }
        lds_copy_wait();
        for (int kk = 0; kk < 64; kk++) {
            int t = cs + kk;
            int dist = q - t;
            if (t >= 0 && dist >= 0 && dist < WINC) {
                float s = 0.f;
#pragma unroll 8
                for (int d = 0; d < DHH; d++) s += qv[d] * Ks[kk][d];
                s *= 0.125f;   // 1/sqrt(64)
                float nm = fmaxf(m, s);
                float sc = __expf(m - nm);
                float p  = __expf(s - nm);
                l = l * sc + p;
#pragma unroll 8
                for (int d = 0; d < DHH; d++) accv[d] = accv[d] * sc + p * Vs[kk][d];
                m = nm;
            }
        }
        __syncthreads();
    }
    float inv = 1.0f / l;
#pragma unroll 8
    for (int d = 0; d < DHH; d++)
        o[((size_t)b * TT + q) * (HH * DHH) + h * DHH + d] = accv[d] * inv;
}

// ---------------- delta memory: out = (triu(kq)*decay)@vals + beta*vals ----------------
// grid (T/64, H, B), block 256: thread = (ql = tid%64, dg = tid/64 -> 16 dims)
__global__ __launch_bounds__(256)
void kern_delta(const float* __restrict__ kb, const float* __restrict__ vb,
                const float* __restrict__ qb, const float* __restrict__ beta,
                const float* __restrict__ cum, float* __restrict__ o) {
    __shared__ float Qs[64][64], Ksh[64][64], Vsh[64][64];
    __shared__ float cQ[64], cK[64];
    const int qt = blockIdx.x, h = blockIdx.y, b = blockIdx.z;
    const int ql = threadIdx.x & 63, dg = threadIdx.x >> 6;
    const int q = qt * 64 + ql;
    const size_t hd = (size_t)HH * DHH;

    for (int u = threadIdx.x; u < 64 * 16; u += 256) {
        int rr = u >> 4, dd = (u & 15) * 4;
        lds_copy_b128(&qb[((size_t)b * TT + qt * 64 + rr) * hd + h * DHH + dd], &Qs[rr][dd]);
    }
    if (threadIdx.x < 64)
        cQ[threadIdx.x] = cum[((size_t)b * HH + h) * TT + qt * 64 + threadIdx.x];
    lds_copy_wait();

    float acc[16];
#pragma unroll
    for (int j = 0; j < 16; j++) acc[j] = 0.f;

    for (int kt = qt; kt < TT / 64; kt++) {
        for (int u = threadIdx.x; u < 64 * 16; u += 256) {
            int rr = u >> 4, dd = (u & 15) * 4;
            size_t base = ((size_t)b * TT + kt * 64 + rr) * hd + h * DHH + dd;
            lds_copy_b128(&kb[base], &Ksh[rr][dd]);
            lds_copy_b128(&vb[base], &Vsh[rr][dd]);
        }
        if (threadIdx.x < 64)
            cK[threadIdx.x] = cum[((size_t)b * HH + h) * TT + kt * 64 + threadIdx.x];
        lds_copy_wait();
        for (int kk = 0; kk < 64; kk++) {
            int k = kt * 64 + kk;
            if (k >= q) {
                float s = 0.f;
#pragma unroll 8
                for (int d = 0; d < DHH; d++) s += Qs[ql][d] * Ksh[kk][d];
                s *= 0.125f;
                float ld  = cQ[ql] - cK[kk];
                float dec = __expf(fminf(ld, 0.f));
                float w = s * dec;
#pragma unroll
                for (int j = 0; j < 16; j++) acc[j] += w * Vsh[kk][dg * 16 + j];
            }
        }
        __syncthreads();
    }
    float bq = beta[((size_t)b * HH + h) * TT + q];
#pragma unroll
    for (int j = 0; j < 16; j++) {
        int d = dg * 16 + j;
        size_t idx = ((size_t)b * TT + q) * hd + h * DHH + d;
        o[idx] = acc[j] + bq * vb[idx];
    }
}

// ---------------- STMM sequential entity-slot scan ----------------
__device__ void stmm_quant(const float* __restrict__ src, const float* __restrict__ cb,
                           float* s_dist, float* s_red, int* s_bmin, float* out) {
    const int tid = threadIdx.x;
    const int c = tid >> 2, part = tid & 3;
    float s = 0.f;
    for (int d = part * 192; d < part * 192 + 192; d++) {
        float df = src[d] - cb[(size_t)c * DDIM + d];
        s += df * df;
    }
    s_red[tid] = s; __syncthreads();
    if (part == 0) s_dist[c] = s_red[tid] + s_red[tid + 1] + s_red[tid + 2] + s_red[tid + 3];
    __syncthreads();
    if (tid == 0) {
        float best = s_dist[0]; int bi = 0;
        for (int c2 = 1; c2 < KCB; c2++)
            if (s_dist[c2] < best) { best = s_dist[c2]; bi = c2; }
        *s_bmin = bi;
    }
    __syncthreads();
    const int bi = *s_bmin;
    // straight-through: z + (zq - z) in f32, faithful to reference arithmetic
    for (int d = tid; d < DDIM; d += 256)
        out[d] = src[d] + (cb[(size_t)bi * DDIM + d] - src[d]);
    __syncthreads();
}

__global__ __launch_bounds__(256)
void kern_stmm(const float* __restrict__ x, const int* __restrict__ tag_pos,
               const int* __restrict__ tag_tok, const int* __restrict__ char_id_p,
               const float* __restrict__ Wih, const float* __restrict__ Whh,
               const float* __restrict__ bih, const float* __restrict__ bhh,
               const float* __restrict__ cb, const float* __restrict__ injW,
               float* __restrict__ inj) {
    __shared__ float s_states[ES * DDIM];
    __shared__ float s_x[DDIM];
    __shared__ float s_gi[3 * DDIM];
    __shared__ float s_gh[3 * DDIM];
    __shared__ float s_h[DDIM];
    __shared__ float s_val[DDIM];
    __shared__ float s_dist[KCB];
    __shared__ float s_red[256];
    __shared__ int s_bmin;
    __shared__ int s_active;

    const int b = blockIdx.x, tid = threadIdx.x;
    for (int i = tid; i < ES * DDIM; i += 256) s_states[i] = 0.f;
    if (tid == 0) s_active = 0;
    const int char_id = char_id_p[0];
    __syncthreads();

    for (int s = 0; s < NTG; s++) {
        const int p  = tag_pos[b * NTG + s];
        const int tk = tag_tok[b * NTG + s];
        for (int d = tid; d < DDIM; d += 256)
            s_x[d] = x[((size_t)b * TT + p) * DDIM + d];
        __syncthreads();

        const int active = s_active;
        const bool is_char = (tk == char_id);
        if (is_char) {
            stmm_quant(s_x, cb, s_dist, s_red, &s_bmin, s_val);
            const int slot = active & (ES - 1);
            for (int d = tid; d < DDIM; d += 256) s_states[slot * DDIM + d] = s_val[d];
            if (tid == 0) s_active = active + 1;
        } else if (active > 0) {
            const int slot = (active - 1) & (ES - 1);
            const float* st = s_states + slot * DDIM;
            for (int r = tid; r < 3 * DDIM; r += 256) {
                float a = bih[r], c2 = bhh[r];
                const float* wi = Wih + (size_t)r * DDIM;
                const float* wh = Whh + (size_t)r * DDIM;
                for (int d = 0; d < DDIM; d++) { a += wi[d] * s_x[d]; c2 += wh[d] * st[d]; }
                s_gi[r] = a; s_gh[r] = c2;
            }
            __syncthreads();
            for (int d = tid; d < DDIM; d += 256) {
                float r = 1.0f / (1.0f + __expf(-(s_gi[d] + s_gh[d])));
                float z = 1.0f / (1.0f + __expf(-(s_gi[DDIM + d] + s_gh[DDIM + d])));
                float n = tanhf(s_gi[2 * DDIM + d] + r * s_gh[2 * DDIM + d]);
                s_h[d] = (1.0f - z) * n + z * st[d];
            }
            __syncthreads();
            stmm_quant(s_h, cb, s_dist, s_red, &s_bmin, s_val);
            for (int d = tid; d < DDIM; d += 256) s_states[slot * DDIM + d] = s_val[d];
        }
        __syncthreads();

        const int actn = s_active;
        if (actn > 0) {
            const int si = (actn - 1) & (ES - 1);
            const float* st = s_states + si * DDIM;
            for (int r = tid; r < DDIM; r += 256) {
                float a = 0.f;
                const float* wr = injW + (size_t)r * DDIM;
                for (int d = 0; d < DDIM; d++) a += wr[d] * st[d];
                inj[((size_t)b * TT + p) * DDIM + r] = a;
            }
        } else {
            for (int r = tid; r < DDIM; r += 256)
                inj[((size_t)b * TT + p) * DDIM + r] = 0.f;
        }
        __syncthreads();
    }
}

// =================================================================
// host orchestration
// =================================================================
static inline dim3 ew_grid(size_t n) {
    size_t bl = (n + 255) / 256;
    if (bl > 8192) bl = 8192;
    return dim3((unsigned)bl);
}

extern "C" void kernel_launch(void* const* d_in, const int* in_sizes, int n_in,
                              void* d_out, int out_size, void* d_ws, size_t ws_size,
                              hipStream_t stream) {
    (void)in_sizes; (void)n_in; (void)out_size; (void)ws_size;
    const float* emb       = (const float*)d_in[0];
    const float* head_W    = (const float*)d_in[1];
    const float* lnf_w     = (const float*)d_in[2];
    const float* ln1_w     = (const float*)d_in[3];
    const float* lnd_w     = (const float*)d_in[4];
    const float* ln2_w     = (const float*)d_in[5];
    const float* qkv_W     = (const float*)d_in[6];
    const float* attn_outW = (const float*)d_in[7];
    const float* k_W       = (const float*)d_in[8];
    const float* v_W       = (const float*)d_in[9];
    const float* q_W       = (const float*)d_in[10];
    const float* beta_W    = (const float*)d_in[11];
    const float* memout_W  = (const float*)d_in[12];
    const float* cg_W      = (const float*)d_in[13];
    const float* co_W      = (const float*)d_in[14];
    const float* Wg_W      = (const float*)d_in[15];
    const float* Wu_W      = (const float*)d_in[16];
    const float* Wo_W      = (const float*)d_in[17];
    const float* gru_Wih   = (const float*)d_in[18];
    const float* gru_Whh   = (const float*)d_in[19];
    const float* gru_bih   = (const float*)d_in[20];
    const float* gru_bhh   = (const float*)d_in[21];
    const float* codebook  = (const float*)d_in[22];
    const float* inj_W     = (const float*)d_in[23];
    const int*   tokens    = (const int*)d_in[24];
    const int*   tag_pos   = (const int*)d_in[25];
    const int*   tag_tok   = (const int*)d_in[26];
    const int*   char_id   = (const int*)d_in[27];
    float* out = (float*)d_out;

    const size_t BT  = (size_t)BB * TT;         // 2048
    const size_t BTD = BT * DDIM;               // activation size

    float* ws = (float*)d_ws;
    size_t off = 0;
    auto alloc = [&](size_t n) { float* p = ws + off; off += (n + 63) & ~(size_t)63; return p; };

    float* xbuf  = alloc(BTD);
    float* inj   = alloc(BTD);
    float* h1    = alloc(BTD);
    float* h2    = alloc(BTD);
    float* obuf  = alloc(BTD);
    float* loc   = alloc(BTD);
    float* glob  = alloc(BTD);
    float* gate  = alloc(BTD);
    float* kbuf  = alloc(BTD);
    float* vbuf  = alloc(BTD);
    float* qbuf  = alloc(BTD);
    float* qkv   = alloc(BT * 3 * HH * DHH);
    float* gbuf  = alloc(BT * DFFC);
    float* ubuf  = alloc(BT * DFFC);
    float* betab = alloc((size_t)BB * HH * TT);
    float* cumb  = alloc((size_t)BB * HH * TT);

    auto gemm = [&](const float* A, const float* W2, float* C, const float* Res,
                    int M, int N, int Kd, int epi) {
        dim3 g((unsigned)(N / 128), (unsigned)(M / 128));
        kern_gemm<<<g, 256, 0, stream>>>(A, W2, C, Res, M, N, Kd, epi);
    };

    // ---- embed + STMM injection ----
    kern_embed<<<ew_grid(BTD), 256, 0, stream>>>(emb, tokens, xbuf, BTD);
    kern_zero<<<ew_grid(BTD), 256, 0, stream>>>(inj, BTD);
    kern_stmm<<<dim3(BB), 256, 0, stream>>>(xbuf, tag_pos, tag_tok, char_id,
                                            gru_Wih, gru_Whh, gru_bih, gru_bhh,
                                            codebook, inj_W, inj);
    kern_add<<<ew_grid(BTD), 256, 0, stream>>>(xbuf, inj, BTD);

    const size_t DD2 = (size_t)DDIM * DDIM;
    for (int l = 0; l < LL; l++) {
        // norms
        kern_rms<<<dim3((unsigned)BT), 256, 0, stream>>>(xbuf, ln1_w + (size_t)l * DDIM, h1);
        kern_rms<<<dim3((unsigned)BT), 256, 0, stream>>>(xbuf, lnd_w + (size_t)l * DDIM, h2);

        // h1 path: qkv + gate
        gemm(h1, qkv_W + (size_t)l * 3 * HH * DHH * DDIM, qkv, nullptr,
             (int)BT, 3 * HH * DHH, DDIM, 0);
        gemm(h1, cg_W + (size_t)l * DD2, gate, nullptr, (int)BT, DDIM, DDIM, 1); // sigmoid

        // sliding-window attention -> obuf, then out-proj -> loc
        kern_swa<<<dim3(TT / 64, HH, BB), 64, 0, stream>>>(qkv, obuf);
        gemm(obuf, attn_outW + (size_t)l * DD2, loc, nullptr, (int)BT, DDIM, DDIM, 0);

        // h2 path: delta memory
        gemm(h2, k_W + (size_t)l * DD2, kbuf, nullptr, (int)BT, DDIM, DDIM, 0);
        gemm(h2, v_W + (size_t)l * DD2, vbuf, nullptr, (int)BT, DDIM, DDIM, 0);
        gemm(h2, q_W + (size_t)l * DD2, qbuf, nullptr, (int)BT, DDIM, DDIM, 0);
        kern_beta<<<dim3((unsigned)BT), 256, 0, stream>>>(h2, beta_W + (size_t)l * HH * DDIM, betab);
        kern_cumsum<<<dim3(1), 32, 0, stream>>>(betab, cumb);
        kern_l2n<<<ew_grid(BT * HH), 256, 0, stream>>>(kbuf, BT * HH);
        kern_l2n<<<ew_grid(BT * HH), 256, 0, stream>>>(qbuf, BT * HH);
        kern_delta<<<dim3(TT / 64, HH, BB), 256, 0, stream>>>(kbuf, vbuf, qbuf, betab, cumb, obuf);
        gemm(obuf, memout_W + (size_t)l * DD2, glob, nullptr, (int)BT, DDIM, DDIM, 0);

        // combine + residual through co
        kern_mix<<<ew_grid(BTD), 256, 0, stream>>>(loc, gate, glob, BTD);
        gemm(loc, co_W + (size_t)l * DD2, xbuf, xbuf, (int)BT, DDIM, DDIM, 0); // x += mix@co^T

        // MLP
        kern_rms<<<dim3((unsigned)BT), 256, 0, stream>>>(xbuf, ln2_w + (size_t)l * DDIM, h1);
        gemm(h1, Wg_W + (size_t)l * DFFC * DDIM, gbuf, nullptr, (int)BT, DFFC, DDIM, 0);
        gemm(h1, Wu_W + (size_t)l * DFFC * DDIM, ubuf, nullptr, (int)BT, DFFC, DDIM, 0);
        kern_silu_mul<<<ew_grid(BT * DFFC), 256, 0, stream>>>(gbuf, ubuf, BT * DFFC);
        gemm(gbuf, Wo_W + (size_t)l * DDIM * DFFC, xbuf, xbuf, (int)BT, DDIM, DFFC, 0);
    }

    // ---- final norm + LM head ----
    kern_rms<<<dim3((unsigned)BT), 256, 0, stream>>>(xbuf, lnf_w, h1);
    gemm(h1, head_W, out, nullptr, (int)BT, VV, DDIM, 0);
}
